// ReasoningCore_75874892251911
// MI455X (gfx1250) — compile-verified, matrix operand order checked by reading, not run on hardware
//
#include <hip/hip_runtime.h>
#include <hip/hip_bf16.h>
#include <math.h>

#define BATCH  131072
#define DIM    768
#define H1     256
#define BN     64
#define NC     32
#define NSTEPS 8

typedef __attribute__((ext_vector_type(16))) _Float16 v16h;
typedef __attribute__((ext_vector_type(8)))  _Float16 v8h;
typedef __attribute__((ext_vector_type(8)))  float    v8f;
typedef __attribute__((ext_vector_type(4)))  float    v4f;

// ---- packed f16 weight image offsets (in halfs), all multiples of 8 ----
#define OFF_W1 0          // [256,768]
#define OFF_W2 196608     // [64,256]
#define OFF_RA 212992     // [32,64]  (ternary)
#define OFF_WA 215040     // [32,64]  (ternary)
#define OFF_WV 217088     // [64,64]  (ternary)
#define OFF_MM 221184     // [64,128]
#define OFF_WR 229376     // [64,192]
#define OFF_WZ 241664     // [64,192]
#define OFF_WN 253952     // [64,192]
#define OFF_D1 266240     // [64,64]
#define OFF_D2 270336     // [2,64]
#define OFF_END 270464

// ---------------- WMMA helpers (CDNA5 16x16x32 f16 -> f32) ----------------
__device__ inline v8f wmma_f16(v16h a, v16h b, v8f c) {
  return __builtin_amdgcn_wmma_f32_16x16x32_f16(false, a, false, b, (short)0, c,
                                                false, false);
}

// A fragment (16x32, f16) built from a row-major f32 tile; per ISA layout:
// lane<16: M=lane, K = {0..7, 16..23}; lane>=16: M=lane-16, K = {8..15, 24..31}
__device__ inline v16h a_frag_f32(const float* __restrict__ base, int ld, int k0, int lane) {
  int m  = lane & 15;
  int kb = (lane & 16) ? 8 : 0;
  const float* r = base + m * ld + k0 + kb;
  v4f x0 = *(const v4f*)(r);
  v4f x1 = *(const v4f*)(r + 4);
  v4f x2 = *(const v4f*)(r + 16);
  v4f x3 = *(const v4f*)(r + 20);
  v16h a;
#pragma unroll
  for (int i = 0; i < 4; ++i) {
    a[i]      = (_Float16)x0[i];
    a[4 + i]  = (_Float16)x1[i];
    a[8 + i]  = (_Float16)x2[i];
    a[12 + i] = (_Float16)x3[i];
  }
  return a;
}

// A fragment from row-major f16 (LDS staging buffers)
__device__ inline v16h a_frag_f16(const _Float16* __restrict__ base, int ld, int k0, int lane) {
  int m  = lane & 15;
  int kb = (lane & 16) ? 8 : 0;
  const _Float16* r = base + m * ld + k0 + kb;
  v8h x0 = *(const v8h*)(r);
  v8h x1 = *(const v8h*)(r + 16);
  v16h a;
#pragma unroll
  for (int i = 0; i < 8; ++i) { a[i] = x0[i]; a[8 + i] = x1[i]; }
  return a;
}

// B fragment (32x16) from weight stored [N,K] row-major (i.e. torch Linear layout)
__device__ inline v16h b_frag_nk(const _Float16* __restrict__ w, int K, int k0, int ntile, int lane) {
  int n  = (lane & 15) + ntile * 16;
  int kb = (lane & 16) ? 8 : 0;
  const _Float16* r = w + n * K + k0 + kb;
  v8h x0 = *(const v8h*)(r);
  v8h x1 = *(const v8h*)(r + 16);
  v16h b;
#pragma unroll
  for (int i = 0; i < 8; ++i) { b[i] = x0[i]; b[8 + i] = x1[i]; }
  return b;
}

// B fragment from [K,N] row-major f16 (cells in LDS), strided gathers
__device__ inline v16h b_frag_km(const _Float16* __restrict__ bkm, int ldn, int ntile, int lane) {
  int n  = (lane & 15) + ntile * 16;
  int kb = (lane & 16) ? 8 : 0;
  v16h b;
#pragma unroll
  for (int j = 0; j < 16; ++j) {
    int k = ((j >> 3) << 4) + kb + (j & 7);
    b[j]  = bkm[k * ldn + n];
  }
  return b;
}

// ---------------- cross-lane row reductions for the WMMA C layout ----------------
__device__ inline float hsum16(float v) {
  v += __shfl_xor(v, 1, 32);
  v += __shfl_xor(v, 2, 32);
  v += __shfl_xor(v, 4, 32);
  v += __shfl_xor(v, 8, 32);
  return v;
}
__device__ inline float hmax16(float v) {
  v = fmaxf(v, __shfl_xor(v, 1, 32));
  v = fmaxf(v, __shfl_xor(v, 2, 32));
  v = fmaxf(v, __shfl_xor(v, 4, 32));
  v = fmaxf(v, __shfl_xor(v, 8, 32));
  return v;
}

__device__ inline float gelu_exact(float v) {
  return 0.5f * v * (1.0f + erff(v * 0.70710678118f));
}

// LayerNorm across NT*16 columns held in C-layout tiles (per-row mean/var via shfl)
template <int NT>
__device__ inline void ln_tiles(v8f* acc, const float* __restrict__ g,
                                const float* __restrict__ be, int lane) {
  const float invn = 1.0f / (16.0f * NT);
  float s[8], q[8];
#pragma unroll
  for (int r = 0; r < 8; ++r) { s[r] = 0.f; q[r] = 0.f; }
#pragma unroll
  for (int t = 0; t < NT; ++t)
#pragma unroll
    for (int r = 0; r < 8; ++r) { float v = acc[t][r]; s[r] += v; q[r] += v * v; }
#pragma unroll
  for (int r = 0; r < 8; ++r) {
    float m   = hsum16(s[r]) * invn;
    float var = hsum16(q[r]) * invn - m * m;
    s[r] = m;
    q[r] = rsqrtf(var + 1e-5f);
  }
  int c0 = lane & 15;
#pragma unroll
  for (int t = 0; t < NT; ++t) {
    float gg = g[t * 16 + c0], bb = be[t * 16 + c0];
#pragma unroll
    for (int r = 0; r < 8; ++r) acc[t][r] = (acc[t][r] - s[r]) * q[r] * gg + bb;
  }
}

// softmax over 32 columns (2 tiles), with bias
__device__ inline void softmax2(v8f* l, const float* __restrict__ bias, int lane) {
  int c0 = lane & 15;
#pragma unroll
  for (int t = 0; t < 2; ++t) {
    float bb = bias[t * 16 + c0];
#pragma unroll
    for (int r = 0; r < 8; ++r) l[t][r] += bb;
  }
  float mx[8], sm[8];
#pragma unroll
  for (int r = 0; r < 8; ++r) mx[r] = hmax16(fmaxf(l[0][r], l[1][r]));
#pragma unroll
  for (int t = 0; t < 2; ++t)
#pragma unroll
    for (int r = 0; r < 8; ++r) l[t][r] = expf(l[t][r] - mx[r]);
#pragma unroll
  for (int r = 0; r < 8; ++r) sm[r] = 1.0f / hsum16(l[0][r] + l[1][r]);
#pragma unroll
  for (int t = 0; t < 2; ++t)
#pragma unroll
    for (int r = 0; r < 8; ++r) l[t][r] *= sm[r];
}

// wa = softmax(A @ waqT + wa_b), wv = tanh(A @ wvqT + wv_b), accumulate column sums
__device__ inline void wa_wv_accum(v16h a0, v16h a1, const _Float16* __restrict__ WH,
                                   const float* __restrict__ wa_b,
                                   const float* __restrict__ wv_b, float* sacc, int lane) {
  v8f zv = {};
  v8f la[2]; la[0] = zv; la[1] = zv;
#pragma unroll
  for (int t = 0; t < 2; ++t) {
    la[t] = wmma_f16(a0, b_frag_nk(WH + OFF_WA, BN, 0, t, lane), la[t]);
    la[t] = wmma_f16(a1, b_frag_nk(WH + OFF_WA, BN, 32, t, lane), la[t]);
  }
  softmax2(la, wa_b, lane);

  v8f lv[4];
#pragma unroll
  for (int t = 0; t < 4; ++t) {
    lv[t] = zv;
    lv[t] = wmma_f16(a0, b_frag_nk(WH + OFF_WV, BN, 0, t, lane), lv[t]);
    lv[t] = wmma_f16(a1, b_frag_nk(WH + OFF_WV, BN, 32, t, lane), lv[t]);
  }
  int c0 = lane & 15;
#pragma unroll
  for (int t = 0; t < 4; ++t) {
    float bb = wv_b[t * 16 + c0];
#pragma unroll
    for (int r = 0; r < 8; ++r) lv[t][r] = tanhf(lv[t][r] + bb);
  }
  // column sums over the 16 rows of this tile
#pragma unroll
  for (int t = 0; t < 2; ++t) {
    float s = 0.f;
#pragma unroll
    for (int r = 0; r < 8; ++r) s += la[t][r];
    s += __shfl_xor(s, 16, 32);
    if (lane < 16) atomicAdd(&sacc[t * 16 + c0], s);
  }
#pragma unroll
  for (int t = 0; t < 4; ++t) {
    float s = 0.f;
#pragma unroll
    for (int r = 0; r < 8; ++r) s += lv[t][r];
    s += __shfl_xor(s, 16, 32);
    if (lane < 16) atomicAdd(&sacc[32 + t * 16 + c0], s);
  }
}

// ---------------- kernels ----------------

// Pack all weights into one f16 image; apply ternary quantization to ra/wa/wv.
__global__ void rc_wcvt_kernel(const float* w1, const float* w2, const float* ra,
                               const float* wa, const float* wv, const float* mm,
                               const float* wr, const float* wz, const float* wn,
                               const float* d1, const float* d2, _Float16* WH) {
  for (int i = blockIdx.x * blockDim.x + threadIdx.x; i < OFF_END;
       i += gridDim.x * blockDim.x) {
    const float* src; int off; bool tern = false;
    if (i < OFF_W2)      { src = w1; off = i - OFF_W1; }
    else if (i < OFF_RA) { src = w2; off = i - OFF_W2; }
    else if (i < OFF_WA) { src = ra; off = i - OFF_RA; tern = true; }
    else if (i < OFF_WV) { src = wa; off = i - OFF_WA; tern = true; }
    else if (i < OFF_MM) { src = wv; off = i - OFF_WV; tern = true; }
    else if (i < OFF_WR) { src = mm; off = i - OFF_MM; }
    else if (i < OFF_WZ) { src = wr; off = i - OFF_WR; }
    else if (i < OFF_WN) { src = wz; off = i - OFF_WZ; }
    else if (i < OFF_D1) { src = wn; off = i - OFF_WN; }
    else if (i < OFF_D2) { src = d1; off = i - OFF_D1; }
    else                 { src = d2; off = i - OFF_D2; }
    float v = src[off];
    if (tern) v = (fabsf(v) > 0.1f) ? (v > 0.f ? 1.f : -1.f) : 0.f;
    WH[i] = (_Float16)v;
  }
}

__global__ void rc_zero_acc_kernel(float* acc) {
  int i = blockIdx.x * blockDim.x + threadIdx.x;
  if (i < 9 * 96) acc[i] = 0.f;
}

// Encoder: x[B,768] -> GELU/LN(256) -> LN(64) = z; also zero h.
__global__ __launch_bounds__(128, 1) void rc_enc_kernel(
    const float* __restrict__ x, const float* __restrict__ b1,
    const float* __restrict__ g1, const float* __restrict__ be1,
    const float* __restrict__ b2, const float* __restrict__ g2,
    const float* __restrict__ be2, const _Float16* __restrict__ WH,
    float* __restrict__ z, float* __restrict__ h) {
  __shared__ __align__(16) _Float16 stage[4][16 * H1];
  int lane = threadIdx.x & 31, wave = threadIdx.x >> 5;
  int rb = (blockIdx.x * 4 + wave) * 16;
  int c0 = lane & 15, ro = (lane & 16) ? 8 : 0;
  const float* xr = x + (size_t)rb * DIM;
  v8f zv = {};

  v8f acc[16];
#pragma unroll
  for (int t = 0; t < 16; ++t) acc[t] = zv;
  for (int ks = 0; ks < DIM / 32; ++ks) {
    v16h a = a_frag_f32(xr, DIM, ks * 32, lane);
#pragma unroll
    for (int t = 0; t < 16; ++t)
      acc[t] = wmma_f16(a, b_frag_nk(WH + OFF_W1, DIM, ks * 32, t, lane), acc[t]);
  }
#pragma unroll
  for (int t = 0; t < 16; ++t) {
    float bb = b1[t * 16 + c0];
#pragma unroll
    for (int r = 0; r < 8; ++r) acc[t][r] = gelu_exact(acc[t][r] + bb);
  }
  ln_tiles<16>(acc, g1, be1, lane);

  _Float16* st = stage[wave];
#pragma unroll
  for (int t = 0; t < 16; ++t)
#pragma unroll
    for (int r = 0; r < 8; ++r)
      st[(ro + r) * H1 + t * 16 + c0] = (_Float16)acc[t][r];
  __syncthreads();

  v8f a2[4];
#pragma unroll
  for (int t = 0; t < 4; ++t) a2[t] = zv;
  for (int ks = 0; ks < H1 / 32; ++ks) {
    v16h a = a_frag_f16(st, H1, ks * 32, lane);
#pragma unroll
    for (int t = 0; t < 4; ++t)
      a2[t] = wmma_f16(a, b_frag_nk(WH + OFF_W2, H1, ks * 32, t, lane), a2[t]);
  }
#pragma unroll
  for (int t = 0; t < 4; ++t) {
    float bb = b2[t * 16 + c0];
#pragma unroll
    for (int r = 0; r < 8; ++r) a2[t][r] += bb;
  }
  ln_tiles<4>(a2, g2, be2, lane);

  float* zr = z + (size_t)rb * BN;
  float* hr = h + (size_t)rb * BN;
#pragma unroll
  for (int t = 0; t < 4; ++t)
#pragma unroll
    for (int r = 0; r < 8; ++r) {
      zr[(ro + r) * BN + t * 16 + c0] = a2[t][r];
      hr[(ro + r) * BN + t * 16 + c0] = 0.f;
    }
}

// Initial memory write partial sums: softmax(z@waqT+b), tanh(z@wvqT+b) means.
__global__ __launch_bounds__(128, 1) void rc_init_partial_kernel(
    const _Float16* __restrict__ WH, const float* __restrict__ z,
    const float* __restrict__ wa_b, const float* __restrict__ wv_b,
    float* __restrict__ accG) {
  __shared__ float sacc[96];
  if (threadIdx.x < 96) sacc[threadIdx.x] = 0.f;
  __syncthreads();
  int lane = threadIdx.x & 31, wave = threadIdx.x >> 5;
  int rb = (blockIdx.x * 4 + wave) * 16;
  const float* zr = z + (size_t)rb * BN;
  v16h a0 = a_frag_f32(zr, BN, 0, lane);
  v16h a1 = a_frag_f32(zr, BN, 32, lane);
  wa_wv_accum(a0, a1, WH, wa_b, wv_b, sacc, lane);
  __syncthreads();
  if (threadIdx.x < 96) atomicAdd(&accG[threadIdx.x], sacc[threadIdx.x]);
}

__global__ void rc_cells_init_kernel(const float* __restrict__ acc, float* __restrict__ cells) {
  int j = threadIdx.x;  // 64 threads
  float sv = acc[32 + j] * (1.0f / BATCH);
#pragma unroll
  for (int c = 0; c < NC; ++c) cells[c * BN + j] = (acc[c] * (1.0f / BATCH)) * sv;
}

__global__ void rc_cells_update_kernel(const float* __restrict__ acc, float* __restrict__ cells) {
  int c = threadIdx.x;  // 32 threads
  float sa = acc[c] * (1.0f / BATCH);
  float row[BN];
  float n2 = 0.f;
#pragma unroll
  for (int j = 0; j < BN; ++j) {
    float v = cells[c * BN + j] + sa * (acc[32 + j] * (1.0f / BATCH));
    row[j] = v;
    n2 += v * v;
  }
  float nrm = sqrtf(n2);
  float scl = (nrm > 1.f) ? (1.f / nrm) : 1.f;
#pragma unroll
  for (int j = 0; j < BN; ++j) cells[c * BN + j] = row[j] * scl;
}

// One GRU/memory step for 16 rows per wave; all five GEMMs via WMMA.
__global__ __launch_bounds__(128, 1) void rc_step_kernel(
    const _Float16* __restrict__ WH, const float* __restrict__ z,
    float* __restrict__ h, const float* __restrict__ cells,
    const float* __restrict__ ra_b, const float* __restrict__ mm_b,
    const float* __restrict__ mn_g, const float* __restrict__ mn_be,
    const float* __restrict__ wr_b, const float* __restrict__ wz_b,
    const float* __restrict__ wn_b, const float* __restrict__ hn_g,
    const float* __restrict__ hn_be, const float* __restrict__ wa_b,
    const float* __restrict__ wv_b, float* __restrict__ accG, int last) {
  __shared__ __align__(16) _Float16 cellsh[NC * BN];
  __shared__ __align__(16) _Float16 bufA[4][16 * BN];
  __shared__ __align__(16) _Float16 bufB[4][16 * BN];
  __shared__ __align__(16) _Float16 bufR[4][16 * NC];
  __shared__ float sacc[96];

  for (int i = threadIdx.x; i < NC * BN; i += blockDim.x) cellsh[i] = (_Float16)cells[i];
  if (threadIdx.x < 96) sacc[threadIdx.x] = 0.f;
  __syncthreads();

  int lane = threadIdx.x & 31, wave = threadIdx.x >> 5;
  int rb = (blockIdx.x * 4 + wave) * 16;
  int c0 = lane & 15, ro = (lane & 16) ? 8 : 0;
  const float* zr = z + (size_t)rb * BN;
  float* hr = h + (size_t)rb * BN;
  _Float16* bA = bufA[wave];
  _Float16* bB = bufB[wave];
  _Float16* bR = bufR[wave];
  v8f zv = {};

  // --- ra = softmax(h @ raqT + ra_b) ---
  v16h ha0 = a_frag_f32(hr, BN, 0, lane);
  v16h ha1 = a_frag_f32(hr, BN, 32, lane);
  v8f ra[2]; ra[0] = zv; ra[1] = zv;
#pragma unroll
  for (int t = 0; t < 2; ++t) {
    ra[t] = wmma_f16(ha0, b_frag_nk(WH + OFF_RA, BN, 0, t, lane), ra[t]);
    ra[t] = wmma_f16(ha1, b_frag_nk(WH + OFF_RA, BN, 32, t, lane), ra[t]);
  }
  softmax2(ra, ra_b, lane);
#pragma unroll
  for (int t = 0; t < 2; ++t)
#pragma unroll
    for (int r = 0; r < 8; ++r) bR[(ro + r) * NC + t * 16 + c0] = (_Float16)ra[t][r];
  __syncthreads();

  // --- readout = ra @ cells ---
  v16h af = a_frag_f16(bR, NC, 0, lane);
  v8f rd[4];
#pragma unroll
  for (int t = 0; t < 4; ++t) { rd[t] = zv; rd[t] = wmma_f16(af, b_frag_km(cellsh, BN, t, lane), rd[t]); }
#pragma unroll
  for (int t = 0; t < 4; ++t)
#pragma unroll
    for (int r = 0; r < 8; ++r) bA[(ro + r) * BN + t * 16 + c0] = (_Float16)rd[t][r];
  __syncthreads();

  // --- mem_out = LN(tanh([h, readout] @ mmT + mm_b)) ---
  v8f mo[4];
#pragma unroll
  for (int t = 0; t < 4; ++t) mo[t] = zv;
#pragma unroll
  for (int ks = 0; ks < 4; ++ks) {
    v16h a;
    if (ks == 0) a = ha0;
    else if (ks == 1) a = ha1;
    else a = a_frag_f16(bA, BN, (ks - 2) * 32, lane);
#pragma unroll
    for (int t = 0; t < 4; ++t)
      mo[t] = wmma_f16(a, b_frag_nk(WH + OFF_MM, 128, ks * 32, t, lane), mo[t]);
  }
#pragma unroll
  for (int t = 0; t < 4; ++t) {
    float bb = mm_b[t * 16 + c0];
#pragma unroll
    for (int r = 0; r < 8; ++r) mo[t][r] = tanhf(mo[t][r] + bb);
  }
  ln_tiles<4>(mo, mn_g, mn_be, lane);
  __syncthreads();  // readout consumed; reuse bA for mem_out
#pragma unroll
  for (int t = 0; t < 4; ++t)
#pragma unroll
    for (int r = 0; r < 8; ++r) bA[(ro + r) * BN + t * 16 + c0] = (_Float16)mo[t][r];
  __syncthreads();

  // --- r, gz gates over xh = [z, mem_out, h] (192) ---
  v8f rg[4], zg[4];
#pragma unroll
  for (int t = 0; t < 4; ++t) { rg[t] = zv; zg[t] = zv; }
#pragma unroll
  for (int ks = 0; ks < 6; ++ks) {
    v16h a;
    if (ks == 0) a = a_frag_f32(zr, BN, 0, lane);
    else if (ks == 1) a = a_frag_f32(zr, BN, 32, lane);
    else if (ks < 4) a = a_frag_f16(bA, BN, (ks - 2) * 32, lane);
    else a = (ks == 4) ? ha0 : ha1;
#pragma unroll
    for (int t = 0; t < 4; ++t) {
      rg[t] = wmma_f16(a, b_frag_nk(WH + OFF_WR, 192, ks * 32, t, lane), rg[t]);
      zg[t] = wmma_f16(a, b_frag_nk(WH + OFF_WZ, 192, ks * 32, t, lane), zg[t]);
    }
  }
  float hc[4][8];
#pragma unroll
  for (int t = 0; t < 4; ++t) {
    float br = wr_b[t * 16 + c0], bz = wz_b[t * 16 + c0];
#pragma unroll
    for (int r = 0; r < 8; ++r) {
      rg[t][r] = 1.f / (1.f + expf(-(rg[t][r] + br)));
      zg[t][r] = 1.f / (1.f + expf(-(zg[t][r] + bz)));
      hc[t][r] = hr[(ro + r) * BN + t * 16 + c0];
    }
  }
  // stage r*h into bB
#pragma unroll
  for (int t = 0; t < 4; ++t)
#pragma unroll
    for (int r = 0; r < 8; ++r)
      bB[(ro + r) * BN + t * 16 + c0] = (_Float16)(rg[t][r] * hc[t][r]);
  __syncthreads();

  // --- n = tanh([z, mem_out, r*h] @ wnT + wn_b) ---
  v8f ng[4];
#pragma unroll
  for (int t = 0; t < 4; ++t) ng[t] = zv;
#pragma unroll
  for (int ks = 0; ks < 6; ++ks) {
    v16h a;
    if (ks == 0) a = a_frag_f32(zr, BN, 0, lane);
    else if (ks == 1) a = a_frag_f32(zr, BN, 32, lane);
    else if (ks < 4) a = a_frag_f16(bA, BN, (ks - 2) * 32, lane);
    else a = a_frag_f16(bB, BN, (ks - 4) * 32, lane);
#pragma unroll
    for (int t = 0; t < 4; ++t)
      ng[t] = wmma_f16(a, b_frag_nk(WH + OFF_WN, 192, ks * 32, t, lane), ng[t]);
  }
#pragma unroll
  for (int t = 0; t < 4; ++t) {
    float bb = wn_b[t * 16 + c0];
#pragma unroll
    for (int r = 0; r < 8; ++r) ng[t][r] = tanhf(ng[t][r] + bb);
  }

  // --- h = LN((1-gz)*h + gz*n) ---
  v8f hn[4];
#pragma unroll
  for (int t = 0; t < 4; ++t)
#pragma unroll
    for (int r = 0; r < 8; ++r)
      hn[t][r] = (1.f - zg[t][r]) * hc[t][r] + zg[t][r] * ng[t][r];
  ln_tiles<4>(hn, hn_g, hn_be, lane);
  __syncthreads();  // r*h consumed; reuse bB for h_new (f16)
#pragma unroll
  for (int t = 0; t < 4; ++t)
#pragma unroll
    for (int r = 0; r < 8; ++r) {
      float v = hn[t][r];
      hr[(ro + r) * BN + t * 16 + c0] = v;
      bB[(ro + r) * BN + t * 16 + c0] = (_Float16)v;
    }
  __syncthreads();

  // --- memory write address/value partial sums (skipped on last step) ---
  if (!last) {
    v16h na0 = a_frag_f16(bB, BN, 0, lane);
    v16h na1 = a_frag_f16(bB, BN, 32, lane);
    wa_wv_accum(na0, na1, WH, wa_b, wv_b, sacc, lane);
  }
  __syncthreads();
  if (threadIdx.x < 96 && !last) atomicAdd(&accG[threadIdx.x], sacc[threadIdx.x]);
}

// Decoder: LN(GELU(h @ d1T + b1)) @ d2T + b2 -> out[B,2]
__global__ __launch_bounds__(128, 1) void rc_dec_kernel(
    const _Float16* __restrict__ WH, const float* __restrict__ h,
    const float* __restrict__ b1, const float* __restrict__ g,
    const float* __restrict__ be, const float* __restrict__ b2,
    float* __restrict__ out) {
  int lane = threadIdx.x & 31, wave = threadIdx.x >> 5;
  int rb = (blockIdx.x * 4 + wave) * 16;
  int c0 = lane & 15, ro = (lane & 16) ? 8 : 0;
  const float* hr = h + (size_t)rb * BN;
  v8f zv = {};
  v16h a0 = a_frag_f32(hr, BN, 0, lane);
  v16h a1 = a_frag_f32(hr, BN, 32, lane);
  v8f d[4];
#pragma unroll
  for (int t = 0; t < 4; ++t) {
    d[t] = zv;
    d[t] = wmma_f16(a0, b_frag_nk(WH + OFF_D1, BN, 0, t, lane), d[t]);
    d[t] = wmma_f16(a1, b_frag_nk(WH + OFF_D1, BN, 32, t, lane), d[t]);
  }
#pragma unroll
  for (int t = 0; t < 4; ++t) {
    float bb = b1[t * 16 + c0];
#pragma unroll
    for (int r = 0; r < 8; ++r) d[t][r] = gelu_exact(d[t][r] + bb);
  }
  ln_tiles<4>(d, g, be, lane);
#pragma unroll
  for (int o = 0; o < 2; ++o) {
    float p[8];
#pragma unroll
    for (int r = 0; r < 8; ++r) p[r] = 0.f;
#pragma unroll
    for (int t = 0; t < 4; ++t) {
      float w = (float)WH[OFF_D2 + o * BN + t * 16 + c0];
#pragma unroll
      for (int r = 0; r < 8; ++r) p[r] += d[t][r] * w;
    }
#pragma unroll
    for (int r = 0; r < 8; ++r) p[r] = hsum16(p[r]);
    if (c0 == 0) {
      float bb = b2[o];
#pragma unroll
      for (int r = 0; r < 8; ++r) out[(size_t)(rb + ro + r) * 2 + o] = p[r] + bb;
    }
  }
}

extern "C" void kernel_launch(void* const* d_in, const int* in_sizes, int n_in,
                              void* d_out, int out_size, void* d_ws, size_t ws_size,
                              hipStream_t stream) {
  const float* x       = (const float*)d_in[0];
  const float* enc_w1  = (const float*)d_in[1];
  const float* enc_b1  = (const float*)d_in[2];
  const float* enc_g1  = (const float*)d_in[3];
  const float* enc_be1 = (const float*)d_in[4];
  const float* enc_w2  = (const float*)d_in[5];
  const float* enc_b2  = (const float*)d_in[6];
  const float* enc_g2  = (const float*)d_in[7];
  const float* enc_be2 = (const float*)d_in[8];
  const float* ra_w    = (const float*)d_in[9];
  const float* ra_b    = (const float*)d_in[10];
  const float* wa_w    = (const float*)d_in[11];
  const float* wa_b    = (const float*)d_in[12];
  const float* wv_w    = (const float*)d_in[13];
  const float* wv_b    = (const float*)d_in[14];
  const float* mm_w    = (const float*)d_in[15];
  const float* mm_b    = (const float*)d_in[16];
  const float* mn_g    = (const float*)d_in[17];
  const float* mn_be   = (const float*)d_in[18];
  const float* wr_w    = (const float*)d_in[19];
  const float* wr_b    = (const float*)d_in[20];
  const float* wz_w    = (const float*)d_in[21];
  const float* wz_b    = (const float*)d_in[22];
  const float* wn_w    = (const float*)d_in[23];
  const float* wn_b    = (const float*)d_in[24];
  const float* hn_g    = (const float*)d_in[25];
  const float* hn_be   = (const float*)d_in[26];
  const float* dec_w1  = (const float*)d_in[27];
  const float* dec_b1  = (const float*)d_in[28];
  const float* dec_g   = (const float*)d_in[29];
  const float* dec_be  = (const float*)d_in[30];
  const float* dec_w2  = (const float*)d_in[31];
  const float* dec_b2  = (const float*)d_in[32];

  char* ws = (char*)d_ws;
  const size_t WH_BYTES = (size_t)1 << 20;
  _Float16* WH  = (_Float16*)ws;
  float* z      = (float*)(ws + WH_BYTES);
  float* h      = (float*)(ws + WH_BYTES + (size_t)BATCH * BN * sizeof(float));
  float* cells  = (float*)(ws + WH_BYTES + 2ull * BATCH * BN * sizeof(float));
  float* acc    = cells + NC * BN;  // 9*96 floats: [0]=init, [1+s]=step s

  const int NBLK = BATCH / 64;  // 4 waves * 16 rows per 128-thread block

  rc_wcvt_kernel<<<256, 256, 0, stream>>>(enc_w1, enc_w2, ra_w, wa_w, wv_w, mm_w,
                                          wr_w, wz_w, wn_w, dec_w1, dec_w2, WH);
  rc_zero_acc_kernel<<<1, 1024, 0, stream>>>(acc);
  rc_enc_kernel<<<NBLK, 128, 0, stream>>>(x, enc_b1, enc_g1, enc_be1, enc_b2,
                                          enc_g2, enc_be2, WH, z, h);
  rc_init_partial_kernel<<<NBLK, 128, 0, stream>>>(WH, z, wa_b, wv_b, acc);
  rc_cells_init_kernel<<<1, 64, 0, stream>>>(acc, cells);
  for (int s = 0; s < NSTEPS; ++s) {
    rc_step_kernel<<<NBLK, 128, 0, stream>>>(WH, z, h, cells, ra_b, mm_b, mn_g,
                                             mn_be, wr_b, wz_b, wn_b, hn_g, hn_be,
                                             wa_b, wv_b, acc + (1 + s) * 96,
                                             (s == NSTEPS - 1) ? 1 : 0);
    if (s < NSTEPS - 1)
      rc_cells_update_kernel<<<1, 32, 0, stream>>>(acc + (1 + s) * 96, cells);
  }
  rc_dec_kernel<<<NBLK, 128, 0, stream>>>(WH, h, dec_b1, dec_g, dec_be, dec_b2,
                                          (float*)d_out);
}
